// MultiHeadAttention_13073880449045
// MI455X (gfx1250) — matrix operand order repaired and checked by reading, not compiled, not run on hardware
//
#include <hip/hip_runtime.h>

// ---------------------------------------------------------------------------
// MHA forward for B=4, S=2048, D=768, H=12, HD=64 on gfx1250 (MI455X).
// bf16 WMMA (v_wmma_f32_16x16x32_bf16), f32 accumulation, flash-attention
// (online softmax, no 805MB score tensor). K/V tiles staged into LDS with the
// Tensor Data Mover (tensor_load_to_lds + s_wait_tensorcnt); working set
// (~75MB) lives in the 192MB L2.
// ---------------------------------------------------------------------------

typedef __attribute__((ext_vector_type(16))) __bf16          v16bf;
typedef __attribute__((ext_vector_type(8)))  float           v8f;
typedef __attribute__((ext_vector_type(8)))  unsigned short  v8u16;
typedef __attribute__((ext_vector_type(4)))  unsigned int    v4u32;
typedef __attribute__((ext_vector_type(8)))  int             v8i32;
typedef __attribute__((ext_vector_type(4)))  int             v4i32;

struct U16x16 { v8u16 lo, hi; };

#define NB 4
#define NS 2048
#define ND 768
#define NH 12
#define HD 64

static constexpr size_t HB_ELEMS  = (size_t)NB * NS * ND;          // 6291456
static constexpr size_t WT_ELEMS  = (size_t)3 * ND * ND;           // 1769472
static constexpr size_t QKV_HEAD  = (size_t)NB * NH * NS * HD;     // per Q/K/V
// workspace (ushort): [Hb | WT | QKV(3x)]

static __device__ __forceinline__ unsigned short f2bf(float f) {
  unsigned u = __float_as_uint(f);
  u += 0x7FFFu + ((u >> 16) & 1u);           // round-to-nearest-even
  return (unsigned short)(u >> 16);
}

static __device__ __forceinline__ v16bf ld_frag16(const unsigned short* p) {
  U16x16 u;
  u.lo = *(const v8u16*)(p);
  u.hi = *(const v8u16*)(p + 8);
  return __builtin_bit_cast(v16bf, u);
}

static __device__ __forceinline__ v16bf ld_frag2x8(const unsigned short* p0,
                                                   const unsigned short* p1) {
  U16x16 u;
  u.lo = *(const v8u16*)(p0);
  u.hi = *(const v8u16*)(p1);
  return __builtin_bit_cast(v16bf, u);
}

static __device__ __forceinline__ v8f wmma_bf16(v16bf a, v16bf b, v8f c) {
  return __builtin_amdgcn_wmma_f32_16x16x32_bf16(
      /*neg_a=*/false, a, /*neg_b=*/false, b,
      /*c_mod=*/(short)0, c, /*reuse_a=*/false, /*reuse_b=*/false);
}

// ---------------------------------------------------------------------------
// Tensor Data Mover: load a 64x64 bf16 tile (contiguous rows of 64 elements,
// row stride 64 elements) from a [tensor_rows][64] bf16 tensor into LDS.
// Descriptor packing per CDNA5 ISA ch.8 (async tensor).
// ---------------------------------------------------------------------------
#if __has_builtin(__builtin_amdgcn_tensor_load_to_lds)
#define HAVE_TDM 1
static __device__ __forceinline__ void tdm_load_tile64x64_bf16(
    unsigned lds_off, const unsigned short* gptr, unsigned tensor_rows) {
  const unsigned long long ga = (unsigned long long)(size_t)gptr;
  v4u32 g0;
  g0[0] = 1u;                                         // count=1 (valid), user D#
  g0[1] = lds_off;                                    // lds_addr (bytes)
  g0[2] = (unsigned)(ga & 0xFFFFFFFFu);               // global_addr[31:0]
  g0[3] = (unsigned)((ga >> 32) & 0x1FFFFFFu)         // global_addr[56:32]
          | (2u << 30);                               // type=2 ("image")
  v8i32 g1;
  g1[0] = (int)(1u << 16);          // workgroup_mask=0 (no cluster); data_size=1 (2B)
  g1[1] = (int)(64u << 16);         // tensor_dim0 = 64 (low 16 bits)
  g1[2] = (int)(tensor_rows << 16); // tensor_dim0 hi=0 | tensor_dim1[15:0]
  g1[3] = (int)(64u << 16);         // tensor_dim1 hi=0 | tile_dim0 = 64
  g1[4] = 64;                       // tile_dim1 = 64 | tile_dim2 = 0
  g1[5] = 64;                       // tensor_dim0_stride[31:0] = 64 elements
  g1[6] = 0;                        // stride hi | tensor_dim1_stride lo
  g1[7] = 0;
  v4i32 z4 = {0, 0, 0, 0};
#if __clang_major__ >= 23
  v8i32 z8 = {0, 0, 0, 0, 0, 0, 0, 0};
  __builtin_amdgcn_tensor_load_to_lds(g0, g1, z4, z4, z8, 0);
#else
  __builtin_amdgcn_tensor_load_to_lds(g0, g1, z4, z4, 0);
#endif
}
#else
#define HAVE_TDM 0
#endif

// Fallback / non-TDM cooperative copy of a 64x64 ushort tile (contiguous).
static __device__ __forceinline__ void coop_copy_tile64x64(
    unsigned short* __restrict__ dst, const unsigned short* __restrict__ src,
    int tid) {
  const int kk = tid >> 1;              // 0..63 row
  const int dc = (tid & 1) * 32;        // 0 or 32
#pragma unroll
  for (int c = 0; c < 4; ++c)
    *(v8u16*)(dst + kk * 64 + dc + c * 8) =
        *(const v8u16*)(src + (size_t)kk * 64 + dc + c * 8);
}

// ---------------------------------------------------------------------------
// Kernel 1: hidden f32 -> bf16 (row-major [8192][768]), 8 elems/thread
// ---------------------------------------------------------------------------
__global__ __launch_bounds__(256) void cvt_hidden(const float* __restrict__ hs,
                                                  unsigned short* __restrict__ Hb) {
  size_t i = ((size_t)blockIdx.x * 256 + threadIdx.x) * 8;
  if (i >= HB_ELEMS) return;
  float4 a = *(const float4*)(hs + i);
  float4 b = *(const float4*)(hs + i + 4);
  v8u16 o;
  o[0] = f2bf(a.x); o[1] = f2bf(a.y); o[2] = f2bf(a.z); o[3] = f2bf(a.w);
  o[4] = f2bf(b.x); o[5] = f2bf(b.y); o[6] = f2bf(b.z); o[7] = f2bf(b.w);
  *(v8u16*)(Hb + i) = o;
}

// ---------------------------------------------------------------------------
// Kernel 2: W[k][n] f32 -> WT[z][n][k] bf16  (n-major for contiguous B-frags)
// ---------------------------------------------------------------------------
__global__ __launch_bounds__(256) void cvt_weights(const float* __restrict__ Wq,
                                                   const float* __restrict__ Wk,
                                                   const float* __restrict__ Wv,
                                                   unsigned short* __restrict__ WT) {
  int z = blockIdx.y;
  size_t id = (size_t)blockIdx.x * 256 + threadIdx.x;   // < 768*768
  if (id >= (size_t)ND * ND) return;
  int n = (int)(id / ND);
  int k = (int)(id % ND);
  const float* W = (z == 0) ? Wq : ((z == 1) ? Wk : Wv);
  WT[(size_t)z * ND * ND + id] = f2bf(W[(size_t)k * ND + n]);
}

// ---------------------------------------------------------------------------
// Kernel 3: QKV projection GEMM (B reuse hits the 192MB L2; no LDS staging).
//   out[z][r][n] = sum_k Hb[r][k] * W_z[k][n] + bias_z[n]   (Q scaled 1/8)
//   stored bf16 head-major: QKV[z][b][h][s][hd]
// ---------------------------------------------------------------------------
__global__ __launch_bounds__(128) void qkv_gemm(const unsigned short* __restrict__ Hb,
                                                const unsigned short* __restrict__ WT,
                                                const float* __restrict__ bq,
                                                const float* __restrict__ bk,
                                                const float* __restrict__ bv,
                                                unsigned short* __restrict__ QKV) {
  const int z   = blockIdx.z;
  const int nt  = blockIdx.y;
  const int mt  = blockIdx.x;
  const int w     = threadIdx.x >> 5;
  const int lane  = threadIdx.x & 31;
  const int lmod  = lane & 15;
  const int lhalf = lane >> 4;

  const int row0 = mt * 64 + w * 16;
  const int n0   = nt * 64;

  const unsigned short* wz   = WT + (size_t)z * ND * ND;
  const float*          bias = (z == 0) ? bq : ((z == 1) ? bk : bv);
  const float           scale = (z == 0) ? 0.125f : 1.0f;  // 1/sqrt(64) folded into Q

  const unsigned short* arow = Hb + (size_t)(row0 + lmod) * ND;
  const unsigned short* brow[4];
#pragma unroll
  for (int j = 0; j < 4; ++j)
    brow[j] = wz + (size_t)(n0 + j * 16 + lmod) * ND + lhalf * 16;

  v8f c[4] = {};
#pragma unroll 4
  for (int k = 0; k < ND; k += 32) {
    v16bf a = ld_frag16(arow + k + lhalf * 16);
#pragma unroll
    for (int j = 0; j < 4; ++j) {
      v16bf b = ld_frag16(brow[j] + k);
      c[j] = wmma_bf16(a, b, c[j]);
    }
  }

#pragma unroll
  for (int j = 0; j < 4; ++j) {
    const int   n  = n0 + j * 16 + lmod;
    const float bz = bias[n];
    const int   h  = n >> 6;
    const int   hd = n & 63;
#pragma unroll
    for (int r = 0; r < 8; ++r) {
      const int row = row0 + r + lhalf * 8;     // C-layout: M = r + 8*(lane/16)
      const int b   = row >> 11;
      const int s   = row & 2047;
      size_t idx = ((((size_t)z * NB + b) * NH + h) * NS + s) * HD + hd;
      QKV[idx] = f2bf((c[j][r] + bz) * scale);
    }
  }
}

// ---------------------------------------------------------------------------
// Kernel 4: flash attention.
//   grid (32 q-tiles, 12 heads, 4 batch), block 128 (4 waves, 16 q-rows each).
//   K/V tiles DMA'd into LDS by the Tensor Data Mover (wave 0 issues, waits
//   TENSORcnt, barrier releases consumers). Scores via wmma from LDS K tile,
//   online softmax with shfl row-reductions, P through per-wave LDS to build
//   A-frags, ctx += P*V with V transposed LDS->LDS.
// ---------------------------------------------------------------------------
__global__ __launch_bounds__(128) void flash_attn(const unsigned short* __restrict__ QKV,
                                                  float* __restrict__ out) {
  const int qt = blockIdx.x;
  const int h  = blockIdx.y;
  const int b  = blockIdx.z;
  const int tid   = threadIdx.x;
  const int w     = tid >> 5;
  const int lane  = tid & 31;
  const int lmod  = lane & 15;
  const int lhalf = lane >> 4;

  const size_t headoff = (((size_t)b * NH + h) * NS) * HD;
  const unsigned short* Q = QKV + 0 * QKV_HEAD + headoff;
  const unsigned short* K = QKV + 1 * QKV_HEAD + headoff;
  const unsigned short* V = QKV + 2 * QKV_HEAD + headoff;

  __shared__ __align__(16) unsigned short Ks[64][64];     // K tile  [key][d]
  __shared__ __align__(16) unsigned short Vs[64][64];     // V tile  [key][d]
  __shared__ __align__(16) unsigned short Vt[64][72];     // V^T     [d][key]
  __shared__ __align__(16) unsigned short Pl[4][16][64];  // per-wave P tile

  // Q A-fragments (16 rows x 64 d), loaded once. Softmax scale pre-folded.
  const unsigned short* qrow = Q + (size_t)(qt * 64 + w * 16 + lmod) * HD;
  v16bf qa0 = ld_frag16(qrow + 0  + lhalf * 16);  // d 0..31
  v16bf qa1 = ld_frag16(qrow + 32 + lhalf * 16);  // d 32..63

  v8f   acc[4] = {};
  float m[8], l[8];
#pragma unroll
  for (int r = 0; r < 8; ++r) { m[r] = -1e30f; l[r] = 0.0f; }

  for (int kt = 0; kt < NS; kt += 64) {
    __syncthreads();  // all readers of Ks/Vs/Vt from previous tile are done

    // ---- stage K and V tiles into LDS via the Tensor Data Mover ----
#if HAVE_TDM
    if (w == 0) {
      tdm_load_tile64x64_bf16((unsigned)(size_t)&Ks[0][0], K + (size_t)kt * HD, NS);
      tdm_load_tile64x64_bf16((unsigned)(size_t)&Vs[0][0], V + (size_t)kt * HD, NS);
      __builtin_amdgcn_s_wait_tensorcnt(0);
    }
#else
    coop_copy_tile64x64(&Ks[0][0], K + (size_t)kt * HD, tid);
    coop_copy_tile64x64(&Vs[0][0], V + (size_t)kt * HD, tid);
#endif
    __syncthreads();  // K/V tiles visible to all waves

    // ---- LDS->LDS transpose of the V tile: Vt[d][key] ----
    {
      const int kk = tid >> 1;
      const int dc = (tid & 1) * 32;
#pragma unroll
      for (int e = 0; e < 32; ++e) Vt[dc + e][kk] = Vs[kk][dc + e];
    }

    // ---- scores S = Q K^T  (16 x 64 per wave), B-frags from LDS ----
    v8f s[4] = {};
#pragma unroll
    for (int j = 0; j < 4; ++j) {
      const unsigned short* krow = &Ks[j * 16 + lmod][0];
      v16bf kb0 = ld_frag16(krow + 0  + lhalf * 16);  // d 0..31
      v16bf kb1 = ld_frag16(krow + 32 + lhalf * 16);  // d 32..63
      s[j] = wmma_bf16(qa0, kb0, s[j]);
      s[j] = wmma_bf16(qa1, kb1, s[j]);
    }

    // ---- online softmax (row r + 8*half; reduced over the 16 lanes of each half) ----
#pragma unroll
    for (int r = 0; r < 8; ++r) {
      float v = fmaxf(fmaxf(s[0][r], s[1][r]), fmaxf(s[2][r], s[3][r]));
      v = fmaxf(v, __shfl_xor(v, 1));
      v = fmaxf(v, __shfl_xor(v, 2));
      v = fmaxf(v, __shfl_xor(v, 4));
      v = fmaxf(v, __shfl_xor(v, 8));
      const float mn    = fmaxf(m[r], v);
      const float alpha = __expf(m[r] - mn);
      m[r] = mn;
      float rs = 0.0f;
#pragma unroll
      for (int j = 0; j < 4; ++j) {
        const float p = __expf(s[j][r] - mn);
        rs += p;
        Pl[w][r + lhalf * 8][j * 16 + lmod] = f2bf(p);
      }
      rs += __shfl_xor(rs, 1);
      rs += __shfl_xor(rs, 2);
      rs += __shfl_xor(rs, 4);
      rs += __shfl_xor(rs, 8);
      l[r] = l[r] * alpha + rs;
#pragma unroll
      for (int j = 0; j < 4; ++j) acc[j][r] *= alpha;
    }

    __syncthreads();  // Vt writes visible to all waves

    // ---- ctx += P V ----
    const unsigned short* prow = &Pl[w][lmod][0];
    v16bf pa0 = ld_frag16(prow + 0  + lhalf * 16);  // keys 0..31
    v16bf pa1 = ld_frag16(prow + 32 + lhalf * 16);  // keys 32..63
#pragma unroll
    for (int j = 0; j < 4; ++j) {
      const unsigned short* vtrow = &Vt[j * 16 + lmod][0];
      v16bf vb0 = ld_frag16(vtrow + 0  + lhalf * 16);
      v16bf vb1 = ld_frag16(vtrow + 32 + lhalf * 16);
      acc[j] = wmma_bf16(pa0, vb0, acc[j]);
      acc[j] = wmma_bf16(pa1, vb1, acc[j]);
    }
  }

  // ---- normalize and write out[b][s][h*64+hd] (f32) ----
#pragma unroll
  for (int r = 0; r < 8; ++r) l[r] = 1.0f / l[r];
#pragma unroll
  for (int j = 0; j < 4; ++j) {
    const int d = h * 64 + j * 16 + lmod;
#pragma unroll
    for (int r = 0; r < 8; ++r) {
      const int srow = qt * 64 + w * 16 + r + lhalf * 8;
      out[((size_t)b * NS + srow) * ND + d] = acc[j][r] * l[r];
    }
  }
}

// ---------------------------------------------------------------------------
extern "C" void kernel_launch(void* const* d_in, const int* in_sizes, int n_in,
                              void* d_out, int out_size, void* d_ws, size_t ws_size,
                              hipStream_t stream) {
  const float* hs = (const float*)d_in[0];
  const float* Wq = (const float*)d_in[1];
  const float* bq = (const float*)d_in[2];
  const float* Wk = (const float*)d_in[3];
  const float* bk = (const float*)d_in[4];
  const float* Wv = (const float*)d_in[5];
  const float* bv = (const float*)d_in[6];
  float* out = (float*)d_out;

  unsigned short* Hb  = (unsigned short*)d_ws;
  unsigned short* WT  = Hb + HB_ELEMS;
  unsigned short* QKV = WT + WT_ELEMS;

  cvt_hidden<<<dim3((unsigned)(HB_ELEMS / (256 * 8))), dim3(256), 0, stream>>>(hs, Hb);
  cvt_weights<<<dim3((ND * ND + 255) / 256, 3), dim3(256), 0, stream>>>(Wq, Wk, Wv, WT);
  qkv_gemm<<<dim3(NB * NS / 64, ND / 64, 3), dim3(128), 0, stream>>>(Hb, WT, bq, bk, bv, QKV);
  flash_attn<<<dim3(NS / 64, NH, NB), dim3(128), 0, stream>>>(QKV, out);
}
